// GCN_2l_28690381537665
// MI455X (gfx1250) — matrix operand order repaired and checked by reading, not compiled, not run on hardware
//
#include <hip/hip_runtime.h>
#include <hip/hip_bf16.h>

// ---------------- problem constants ----------------
static constexpr int Nn   = 10000;   // nodes
static constexpr int Ee   = 160000;  // edges
static constexpr int INC  = 128;     // in channels
static constexpr int HID  = 128;     // hidden per head (conv1)
static constexpr int HEADS= 4;
static constexpr int D1   = HEADS * HID;  // 512
// 10000 = 625 * 16 exactly; 128 = 4 * 32 exactly; 512 = 8 * 64 exactly.

typedef __attribute__((ext_vector_type(16))) __bf16 v16bf;
typedef __attribute__((ext_vector_type(8)))  __bf16 v8bf;
typedef __attribute__((ext_vector_type(8)))  float  v8f;

// ---------------- helpers ----------------
__device__ inline void atomicMaxF(float* addr, float v) {
  // signed/unsigned split trick; works with -inf initialization
  if (v >= 0.0f) atomicMax((int*)addr, __float_as_int(v));
  else           atomicMin((unsigned int*)addr, __float_as_uint(v));
}

__global__ void cast_f32_to_bf16(const float* __restrict__ s,
                                 __bf16* __restrict__ d, int n) {
  int i = blockIdx.x * blockDim.x + threadIdx.x;
  if (i < n) d[i] = (__bf16)s[i];
}

__global__ void fill_f32(float* __restrict__ p, float v, int n) {
  int i = blockIdx.x * blockDim.x + threadIdx.x;
  if (i < n) p[i] = v;
}

// ---------------- conv1 projections: bf16 WMMA GEMM ----------------
// out[mat][n][col] = sum_k x[n][k] * W[mat][col][k] + bias[mat][col]
// One wave: 16 rows x 64 cols (4 WMMA tiles), K-loop 128 in steps of 32.
__global__ __launch_bounds__(256) void gemm_qkvs1(
    const __bf16* __restrict__ xb,   // [Nn,128] bf16
    const __bf16* __restrict__ wb,   // [4,512,128] bf16
    const float* __restrict__ bq, const float* __restrict__ bk,
    const float* __restrict__ bv, const float* __restrict__ bs,
    __bf16* __restrict__ out)        // [4,Nn,512] bf16
{
  const int mat = blockIdx.y;                       // 0=q 1=k 2=v 3=skip
  const __bf16* W   = wb + (size_t)mat * D1 * INC;
  const float* bias = (mat == 0) ? bq : (mat == 1) ? bk : (mat == 2) ? bv : bs;
  __bf16* O = out + (size_t)mat * (size_t)Nn * D1;

  const int wave = blockIdx.x * (blockDim.x >> 5) + (threadIdx.x >> 5);
  const int lane = threadIdx.x & 31;
  const int rowt = wave >> 3;            // 625 row tiles
  const int colt = wave & 7;             // 8 strips of 64 cols
  const int mbase = rowt * 16;
  const int cbase = colt * 64;

  const int ml   = lane & 15;
  const int kb4  = (lane < 16) ? 0 : 8;    // unused; A fragment also uses kb16
  const int kb16 = (lane < 16) ? 0 : 16;   // B fragment K sub-base

  v8f acc0 = {}, acc1 = {}, acc2 = {}, acc3 = {};
  const __bf16* arow = xb + (size_t)(mbase + ml) * INC;

  for (int k0 = 0; k0 < INC; k0 += 32) {
    // A 16x32 bf16 fragment: two contiguous 16B loads per lane
    v8bf alo = *(const v8bf*)(arow + k0 + kb16);       // K = k0+kb16 .. +7
    v8bf ahi = *(const v8bf*)(arow + k0 + kb16 + 8);   // K = k0+kb16+8 .. +15
    v16bf a;
#pragma unroll
    for (int i = 0; i < 8; ++i) { a[i] = alo[i]; a[8 + i] = ahi[i]; }

#pragma unroll
    for (int t = 0; t < 4; ++t) {
      // B 32x16 fragment: b[j] = W[col][k0+kb16+j]  (contiguous 32B load)
      const __bf16* brow = W + (size_t)(cbase + t * 16 + ml) * INC + k0 + kb16;
      v16bf b = *(const v16bf*)brow;
      v8f& acc = (t == 0) ? acc0 : (t == 1) ? acc1 : (t == 2) ? acc2 : acc3;
      acc = __builtin_amdgcn_wmma_f32_16x16x32_bf16(
          /*neg_a=*/false, a, /*neg_b=*/false, b,
          /*c_mod=*/(short)0, acc, /*reuse_a=*/false, /*reuse_b=*/false);
    }
  }

  const int rhalf = (lane < 16) ? 0 : 8;
#pragma unroll
  for (int t = 0; t < 4; ++t) {
    v8f acc = (t == 0) ? acc0 : (t == 1) ? acc1 : (t == 2) ? acc2 : acc3;
    const int col = cbase + t * 16 + ml;
    const float bvl = bias[col];
#pragma unroll
    for (int r = 0; r < 8; ++r) {
      const int m = mbase + rhalf + r;
      O[(size_t)m * D1 + col] = (__bf16)(acc[r] + bvl);
    }
  }
}

// ---------------- conv1 edge attention ----------------
// One wave per edge. lane: head = lane>>3, 16-channel chunk = (lane&7)*16.
__global__ __launch_bounds__(256) void edge_logits1(
    const __bf16* __restrict__ qb, const __bf16* __restrict__ kb,
    const long long* __restrict__ ei,
    float* __restrict__ logits, float* __restrict__ smax)
{
  const int e = blockIdx.x * 8 + (threadIdx.x >> 5);
  const int lane = threadIdx.x & 31;
  const long long s = ei[e];
  const long long d = ei[(size_t)Ee + e];
  const int head = lane >> 3, seg = lane & 7;
  v16bf qv = *(const v16bf*)(qb + (size_t)d * D1 + head * HID + seg * 16);
  v16bf kv = *(const v16bf*)(kb + (size_t)s * D1 + head * HID + seg * 16);
  float acc = 0.f;
#pragma unroll
  for (int i = 0; i < 16; ++i) acc += (float)qv[i] * (float)kv[i];
  acc += __shfl_xor(acc, 1, 32);
  acc += __shfl_xor(acc, 2, 32);
  acc += __shfl_xor(acc, 4, 32);
  if (seg == 0) {
    const float v = acc * 0.08838834764831845f;  // 1/sqrt(128)
    logits[(size_t)e * HEADS + head] = v;
    atomicMaxF(smax + (size_t)d * HEADS + head, v);
  }
}

__global__ void edge_exp1(float* __restrict__ lg, const long long* __restrict__ ei,
                          const float* __restrict__ smax, float* __restrict__ den)
{
  const int t = blockIdx.x * blockDim.x + threadIdx.x;  // E*4 threads
  const int e = t >> 2, h = t & 3;
  const long long d = ei[(size_t)Ee + e];
  const float ex = __expf(lg[t] - smax[(size_t)d * HEADS + h]);
  lg[t] = ex;
  atomicAdd(den + (size_t)d * HEADS + h, ex);
}

__global__ __launch_bounds__(256) void edge_aggr1(
    const __bf16* __restrict__ vb, const long long* __restrict__ ei,
    const float* __restrict__ ex, const float* __restrict__ den,
    float* __restrict__ out1)
{
  const int e = blockIdx.x * 8 + (threadIdx.x >> 5);
  const int lane = threadIdx.x & 31;
  const long long s = ei[e];
  const long long d = ei[(size_t)Ee + e];
  const int head = lane >> 3, seg = lane & 7;
  const float alpha = ex[(size_t)e * HEADS + head] /
                      fmaxf(den[(size_t)d * HEADS + head], 1e-16f);
  v16bf vv = *(const v16bf*)(vb + (size_t)s * D1 + head * HID + seg * 16);
  float* op = out1 + (size_t)d * D1 + head * HID + seg * 16;
#pragma unroll
  for (int i = 0; i < 16; ++i) atomicAdd(op + i, alpha * (float)vv[i]);
}

// ---------------- relu(out1 + skip) -> bf16 ----------------
__global__ void relu_skip(const float* __restrict__ out1,
                          const __bf16* __restrict__ sk,
                          __bf16* __restrict__ hb)
{
  const int i = blockIdx.x * blockDim.x + threadIdx.x;  // Nn*512 threads
  const float v = out1[i] + (float)sk[i];
  hb[i] = (__bf16)fmaxf(v, 0.0f);
}

// ---------------- conv2 projections (4 outputs each, plain dots) ------------
__global__ void conv2_proj(const __bf16* __restrict__ hb,
    const float* __restrict__ Wq, const float* __restrict__ bq,
    const float* __restrict__ Wk, const float* __restrict__ bk,
    const float* __restrict__ Wv, const float* __restrict__ bv,
    const float* __restrict__ Ws, const float* __restrict__ bs,
    float* __restrict__ q2, float* __restrict__ k2,
    float* __restrict__ v2, float* __restrict__ s2)
{
  const int t = blockIdx.x * blockDim.x + threadIdx.x;  // Nn*16 threads
  const int n = t >> 4, j = t & 15, mat = j >> 2, o = j & 3;
  const float* W; const float* b; float* O;
  if      (mat == 0) { W = Wq; b = bq; O = q2; }
  else if (mat == 1) { W = Wk; b = bk; O = k2; }
  else if (mat == 2) { W = Wv; b = bv; O = v2; }
  else               { W = Ws; b = bs; O = s2; }
  const __bf16* hp = hb + (size_t)n * D1;
  const float*  wp = W  + (size_t)o * D1;
  float acc = b[o];
#pragma unroll 8
  for (int c = 0; c < D1; ++c) acc += (float)hp[c] * wp[c];
  O[(size_t)n * HEADS + o] = acc;
}

// ---------------- conv2 edge attention (ch = 1) ----------------
__global__ void edge_logits2(const float* __restrict__ q2, const float* __restrict__ k2,
                             const long long* __restrict__ ei,
                             float* __restrict__ lg, float* __restrict__ smax)
{
  const int t = blockIdx.x * blockDim.x + threadIdx.x;  // E*4 threads
  const int e = t >> 2, h = t & 3;
  const long long s = ei[e];
  const long long d = ei[(size_t)Ee + e];
  const float v = q2[(size_t)d * HEADS + h] * k2[(size_t)s * HEADS + h]; // /sqrt(1)
  lg[t] = v;
  atomicMaxF(smax + (size_t)d * HEADS + h, v);
}

__global__ void edge_aggr2(const float* __restrict__ ex, const float* __restrict__ den,
                           const float* __restrict__ v2, const long long* __restrict__ ei,
                           float* __restrict__ out2)
{
  const int t = blockIdx.x * blockDim.x + threadIdx.x;  // E*4 threads
  const int e = t >> 2, h = t & 3;
  const long long s = ei[e];
  const long long d = ei[(size_t)Ee + e];
  const float alpha = ex[t] / fmaxf(den[(size_t)d * HEADS + h], 1e-16f);
  atomicAdd(out2 + (size_t)d * HEADS + h, alpha * v2[(size_t)s * HEADS + h]);
}

// ---------------- final: mean_n( (out2+skip) @ Wl^T + bl ) ----------------
__global__ __launch_bounds__(256) void final_reduce(
    const float* __restrict__ out2, const float* __restrict__ s2,
    const float* __restrict__ Wl, float* __restrict__ dout)
{
  __shared__ float sm[256];
  float acc = 0.f;
  for (int n = blockIdx.x * blockDim.x + threadIdx.x; n < Nn;
       n += gridDim.x * blockDim.x) {
    float y = 0.f;
#pragma unroll
    for (int h = 0; h < HEADS; ++h)
      y += (out2[(size_t)n * HEADS + h] + s2[(size_t)n * HEADS + h]) * Wl[h];
    acc += y;
  }
  sm[threadIdx.x] = acc;
  __syncthreads();
  for (int s = 128; s > 0; s >>= 1) {
    if (threadIdx.x < s) sm[threadIdx.x] += sm[threadIdx.x + s];
    __syncthreads();
  }
  if (threadIdx.x == 0) atomicAdd(dout, sm[0]);
}

__global__ void finalize(float* __restrict__ dout, const float* __restrict__ bl) {
  dout[0] = dout[0] / (float)Nn + bl[0];
}

// ---------------- host launcher ----------------
extern "C" void kernel_launch(void* const* d_in, const int* in_sizes, int n_in,
                              void* d_out, int out_size, void* d_ws, size_t ws_size,
                              hipStream_t stream) {
  const float*     x   = (const float*)d_in[0];
  const long long* ei  = (const long long*)d_in[1];   // int64 per reference
  const float *Wq1 = (const float*)d_in[2],  *bq1 = (const float*)d_in[3];
  const float *Wk1 = (const float*)d_in[4],  *bk1 = (const float*)d_in[5];
  const float *Wv1 = (const float*)d_in[6],  *bv1 = (const float*)d_in[7];
  const float *Ws1 = (const float*)d_in[8],  *bs1 = (const float*)d_in[9];
  const float *Wq2 = (const float*)d_in[10], *bq2 = (const float*)d_in[11];
  const float *Wk2 = (const float*)d_in[12], *bk2 = (const float*)d_in[13];
  const float *Wv2 = (const float*)d_in[14], *bv2 = (const float*)d_in[15];
  const float *Ws2 = (const float*)d_in[16], *bs2 = (const float*)d_in[17];
  const float *Wl  = (const float*)d_in[18], *bl  = (const float*)d_in[19];
  float* dout = (float*)d_out;

  // workspace carve-out (deterministic, 256B aligned)
  char* ws = (char*)d_ws;
  size_t off = 0;
  auto carve = [&](size_t bytes) -> char* {
    char* p = ws + off;
    off = (off + bytes + 255) & ~(size_t)255;
    return p;
  };
  __bf16* xb    = (__bf16*)carve((size_t)Nn * INC * 2);
  __bf16* wb    = (__bf16*)carve((size_t)4 * D1 * INC * 2);
  __bf16* qkvs1 = (__bf16*)carve((size_t)4 * Nn * D1 * 2);
  float*  out1  = (float*)carve((size_t)Nn * D1 * 4);
  __bf16* hb    = (__bf16*)carve((size_t)Nn * D1 * 2);
  float*  lg    = (float*)carve((size_t)Ee * HEADS * 4);  // logits/ex, reused by conv2
  float*  smax1 = (float*)carve((size_t)Nn * HEADS * 4);
  float*  den1  = (float*)carve((size_t)Nn * HEADS * 4);
  float*  qkvs2 = (float*)carve((size_t)Nn * 16 * 4);
  float*  smax2 = (float*)carve((size_t)Nn * HEADS * 4);
  float*  den2  = (float*)carve((size_t)Nn * HEADS * 4);
  float*  out2  = (float*)carve((size_t)Nn * HEADS * 4);
  float* q2 = qkvs2, *k2 = qkvs2 + (size_t)Nn * 4,
       * v2 = qkvs2 + (size_t)Nn * 8, *s2 = qkvs2 + (size_t)Nn * 12;
  __bf16* q1b = qkvs1;
  __bf16* k1b = qkvs1 + (size_t)Nn * D1;
  __bf16* v1b = qkvs1 + (size_t)2 * Nn * D1;
  __bf16* s1b = qkvs1 + (size_t)3 * Nn * D1;

  // ---- init accumulators (graph-capture safe) ----
  hipMemsetAsync(out1, 0, (size_t)Nn * D1 * 4, stream);
  hipMemsetAsync(den1, 0, (size_t)Nn * HEADS * 4, stream);
  hipMemsetAsync(den2, 0, (size_t)Nn * HEADS * 4, stream);
  hipMemsetAsync(out2, 0, (size_t)Nn * HEADS * 4, stream);
  hipMemsetAsync(dout, 0, sizeof(float), stream);
  fill_f32<<<(Nn * HEADS + 255) / 256, 256, 0, stream>>>(smax1, -INFINITY, Nn * HEADS);
  fill_f32<<<(Nn * HEADS + 255) / 256, 256, 0, stream>>>(smax2, -INFINITY, Nn * HEADS);

  // ---- casts to bf16 ----
  cast_f32_to_bf16<<<(Nn * INC + 255) / 256, 256, 0, stream>>>(x, xb, Nn * INC);
  const int wsz = D1 * INC;  // 65536
  cast_f32_to_bf16<<<wsz / 256, 256, 0, stream>>>(Wq1, wb + 0 * (size_t)wsz, wsz);
  cast_f32_to_bf16<<<wsz / 256, 256, 0, stream>>>(Wk1, wb + 1 * (size_t)wsz, wsz);
  cast_f32_to_bf16<<<wsz / 256, 256, 0, stream>>>(Wv1, wb + 2 * (size_t)wsz, wsz);
  cast_f32_to_bf16<<<wsz / 256, 256, 0, stream>>>(Ws1, wb + 3 * (size_t)wsz, wsz);

  // ---- conv1: 4 projections via WMMA (625*8 waves, 8 waves/block) ----
  gemm_qkvs1<<<dim3(625, 4), 256, 0, stream>>>(xb, wb, bq1, bk1, bv1, bs1, qkvs1);

  // ---- conv1 attention ----
  edge_logits1<<<Ee / 8, 256, 0, stream>>>(q1b, k1b, ei, lg, smax1);
  edge_exp1<<<(Ee * HEADS) / 256, 256, 0, stream>>>(lg, ei, smax1, den1);
  edge_aggr1<<<Ee / 8, 256, 0, stream>>>(v1b, ei, lg, den1, out1);

  // ---- relu + skip ----
  relu_skip<<<(Nn * D1) / 256, 256, 0, stream>>>(out1, s1b, hb);

  // ---- conv2 projections ----
  conv2_proj<<<(Nn * 16) / 256, 256, 0, stream>>>(hb, Wq2, bq2, Wk2, bk2,
                                                  Wv2, bv2, Ws2, bs2,
                                                  q2, k2, v2, s2);

  // ---- conv2 attention (ch=1), reuse lg buffer ----
  edge_logits2<<<(Ee * HEADS) / 256, 256, 0, stream>>>(q2, k2, ei, lg, smax2);
  edge_exp1<<<(Ee * HEADS) / 256, 256, 0, stream>>>(lg, ei, smax2, den2);
  edge_aggr2<<<(Ee * HEADS) / 256, 256, 0, stream>>>(lg, den2, v2, ei, out2);

  // ---- final linear + mean ----
  final_reduce<<<128, 256, 0, stream>>>(out2, s2, Wl, dout);
  finalize<<<1, 1, 0, stream>>>(dout, bl);
}
